// FusionModel_3736621547772
// MI455X (gfx1250) — matrix operand order repaired and checked by reading, not compiled, not run on hardware
//
#include <hip/hip_runtime.h>
#include <math.h>

// ---------------------------------------------------------------------------
// FusionModel forward for MI455X (gfx1250), wave32.
// GEMMs use split-bf16 (hi/lo) with V_WMMA_F32_16X16X32_BF16, fp32 accumulate:
//   a*b ~= ah*bh + ah*bl + al*bh   (per-term rel. err ~3e-5, fp32-class)
// cross_atten is an exact identity (softmax rows sum to 1) and is elided.
// ---------------------------------------------------------------------------

typedef float  v8f   __attribute__((ext_vector_type(8)));
typedef __bf16 bf16;
typedef bf16   v8bf  __attribute__((ext_vector_type(8)));
typedef bf16   v16bf __attribute__((ext_vector_type(16)));

#define B_ROWS   16384
#define DC       2048
#define DB       768
#define DF       256
#define NF       768
#define EPSF     1e-5f

// LDS tiles: [64 rows][32 k] bf16, row stride 40 bf16 (80 B):
//  - 16B aligned rows (b128 frag loads)
//  - bank(m) = 20*m mod 64 distinct for 16 consecutive rows -> conflict-free
#define T_LD 40

__device__ __forceinline__ v8bf ld8(const bf16* p) { return *(const v8bf*)p; }
__device__ __forceinline__ v16bf cat16(v8bf a, v8bf b) {
  return __builtin_shufflevector(a, b, 0, 1, 2, 3, 4, 5, 6, 7,
                                 8, 9, 10, 11, 12, 13, 14, 15);
}
__device__ __forceinline__ void split_store(bf16* hp, bf16* lp, float v) {
  const bf16 h = (bf16)v;
  *hp = h;
  *lp = (bf16)(v - (float)h);
}

// --- split-bf16 WMMA over one staged 64x32 (A) / 32x64 (B) hi/lo tile set ---
// Wave (wm,wn) owns rows [wm*16,+16) and cols [wn*32,+32) of the 64x64 block.
__device__ __forceinline__ void wmma_tile(const bf16* __restrict__ Ahi,
                                          const bf16* __restrict__ Alo,
                                          const bf16* __restrict__ Bhi,
                                          const bf16* __restrict__ Blo,
                                          int wm, int wn, int lane,
                                          v8f& acc0, v8f& acc1) {
  const int half = lane >> 4, lm = lane & 15;
  // A frag (16x32 bf16): V0-3 = K[8h..8h+8), V4-7 = K[16+8h..16+8h+8)
  const int ao = (wm * 16 + lm) * T_LD + 8 * half;
  const v16bf ah = cat16(ld8(Ahi + ao), ld8(Ahi + ao + 16));
  const v16bf al = cat16(ld8(Alo + ao), ld8(Alo + ao + 16));
  // B frag (32x16 bf16): lane holds col n, V0-3 = K[8h..8h+8), V4-7 = K[16+8h..16+8h+8)
  const int b0 = (wn * 32 + lm) * T_LD + 8 * half;
  const int b1 = (wn * 32 + 16 + lm) * T_LD + 8 * half;
  const v16bf bh0 = cat16(ld8(Bhi + b0), ld8(Bhi + b0 + 16));
  const v16bf bl0 = cat16(ld8(Blo + b0), ld8(Blo + b0 + 16));
  const v16bf bh1 = cat16(ld8(Bhi + b1), ld8(Bhi + b1 + 16));
  const v16bf bl1 = cat16(ld8(Blo + b1), ld8(Blo + b1 + 16));
  acc0 = __builtin_amdgcn_wmma_f32_16x16x32_bf16(false, ah, false, bh0, (short)0, acc0, false, false);
  acc0 = __builtin_amdgcn_wmma_f32_16x16x32_bf16(false, ah, false, bl0, (short)0, acc0, false, false);
  acc0 = __builtin_amdgcn_wmma_f32_16x16x32_bf16(false, al, false, bh0, (short)0, acc0, false, false);
  acc1 = __builtin_amdgcn_wmma_f32_16x16x32_bf16(false, ah, false, bh1, (short)0, acc1, false, false);
  acc1 = __builtin_amdgcn_wmma_f32_16x16x32_bf16(false, ah, false, bl1, (short)0, acc1, false, false);
  acc1 = __builtin_amdgcn_wmma_f32_16x16x32_bf16(false, al, false, bh1, (short)0, acc1, false, false);
}

// C/D layout (§7.12.2): lane l, vgpr r -> row = r + 8*(l>>4), col = l&15.
__device__ __forceinline__ void store_tile(float* __restrict__ out, int ldN,
                                           int rowBase, int colBase,
                                           int wm, int wn, int lane,
                                           const v8f& acc0, const v8f& acc1,
                                           const float* __restrict__ bias) {
  const int half = lane >> 4, lm = lane & 15;
  const int col0 = colBase + wn * 32 + lm;
  const int col1 = col0 + 16;
  const float bv0 = bias[col0];
  const float bv1 = bias[col1];
#pragma unroll
  for (int r = 0; r < 8; ++r) {
    const int row = rowBase + wm * 16 + half * 8 + r;
    out[(size_t)row * ldN + col0] = acc0[r] + bv0;
    out[(size_t)row * ldN + col1] = acc1[r] + bv1;
  }
}

// --- K1: ctx = x_context @ Wc^T + bc   [16384x2048]x[2048x768] --------------
__global__ __launch_bounds__(256) void k_gemm_ctx(const float* __restrict__ X,
                                                  const float* __restrict__ Wc,
                                                  const float* __restrict__ bc,
                                                  float* __restrict__ ctx) {
  __shared__ __align__(16) bf16 Ahi[64 * T_LD];
  __shared__ __align__(16) bf16 Alo[64 * T_LD];
  __shared__ __align__(16) bf16 Bhi[64 * T_LD];
  __shared__ __align__(16) bf16 Blo[64 * T_LD];
  const int t = threadIdx.x, lane = t & 31, wid = t >> 5;
  const int wm = wid & 3, wn = wid >> 2;
  const int tileN = blockIdx.x * 64, tileM = blockIdx.y * 64;
  const int m = t >> 2, kq = (t & 3) << 3;    // staging: 8 elements per thread
  v8f acc0 = {}, acc1 = {};
  for (int kb0 = 0; kb0 < DC; kb0 += 32) {
    __syncthreads();
    const float* ap = X + (size_t)(tileM + m) * DC + kb0 + kq;
    const float* bp = Wc + (size_t)(tileN + m) * DC + kb0 + kq;  // B in [n][k] form
#pragma unroll
    for (int i = 0; i < 8; ++i) {
      split_store(&Ahi[m * T_LD + kq + i], &Alo[m * T_LD + kq + i], ap[i]);
      split_store(&Bhi[m * T_LD + kq + i], &Blo[m * T_LD + kq + i], bp[i]);
    }
    __syncthreads();
    wmma_tile(Ahi, Alo, Bhi, Blo, wm, wn, lane, acc0, acc1);
  }
  store_tile(ctx, NF, tileM, tileN, wm, wn, lane, acc0, acc1, bc);
}

// --- K2: Y[f] = feat_f @ Wa1^T + ba1, f in {ctx, body, face} ----------------
__global__ __launch_bounds__(256) void k_gemm_att(const float* __restrict__ ctx,
                                                  const float* __restrict__ xbody,
                                                  const float* __restrict__ xface,
                                                  const float* __restrict__ Wa1,
                                                  const float* __restrict__ ba1,
                                                  float* __restrict__ Y) {
  __shared__ __align__(16) bf16 Ahi[64 * T_LD];
  __shared__ __align__(16) bf16 Alo[64 * T_LD];
  __shared__ __align__(16) bf16 Bhi[64 * T_LD];
  __shared__ __align__(16) bf16 Blo[64 * T_LD];
  const int t = threadIdx.x, lane = t & 31, wid = t >> 5;
  const int wm = wid & 3, wn = wid >> 2;
  const int tileN = blockIdx.x * 64, tileM = blockIdx.y * 64, f = blockIdx.z;
  const int m = t >> 2, kq = (t & 3) << 3;
  const int row = tileM + m;
  v8f acc0 = {}, acc1 = {};
  for (int kb0 = 0; kb0 < NF; kb0 += 32) {
    __syncthreads();
#pragma unroll
    for (int i = 0; i < 8; ++i) {
      const int k = kb0 + kq + i;
      float v;
      if (f == 0)      v = ctx[(size_t)row * NF + k];
      else if (f == 1) v = xbody[(size_t)row * NF + k];
      else             v = xface[(size_t)row * DF + k / 3];  // repeat(x_face,3)
      split_store(&Ahi[m * T_LD + kq + i], &Alo[m * T_LD + kq + i], v);
    }
    const float* bp = Wa1 + (size_t)(tileN + m) * NF + kb0 + kq;
#pragma unroll
    for (int i = 0; i < 8; ++i)
      split_store(&Bhi[m * T_LD + kq + i], &Blo[m * T_LD + kq + i], bp[i]);
    __syncthreads();
    wmma_tile(Ahi, Alo, Bhi, Blo, wm, wn, lane, acc0, acc1);
  }
  store_tile(Y + (size_t)f * B_ROWS * 128, 128, tileM, tileN, wm, wn, lane,
             acc0, acc1, ba1);
}

// --- K3/K6: per-channel batch mean / rstd (no atomics) ----------------------
__global__ __launch_bounds__(256) void k_stats(const float* __restrict__ X,
                                               int ldc,
                                               float* __restrict__ meanOut,
                                               float* __restrict__ rstdOut) {
  __shared__ float sbuf[256];
  __shared__ float qbuf[256];
  const int c = blockIdx.x, t = threadIdx.x;
  const int f = c / ldc, n = c % ldc;
  const float* p = X + (size_t)f * B_ROWS * ldc + n;
  float s = 0.f, q = 0.f;
  for (int b = t; b < B_ROWS; b += 256) {
    const float v = p[(size_t)b * ldc];
    s += v; q += v * v;
  }
  sbuf[t] = s; qbuf[t] = q;
  __syncthreads();
  for (int off = 128; off > 0; off >>= 1) {
    if (t < off) { sbuf[t] += sbuf[t + off]; qbuf[t] += qbuf[t + off]; }
    __syncthreads();
  }
  if (t == 0) {
    const float mean = sbuf[0] * (1.f / B_ROWS);
    const float var  = qbuf[0] * (1.f / B_ROWS) - mean * mean;
    meanOut[c] = mean;
    rstdOut[c] = rsqrtf(var + EPSF);
  }
}

// --- K4: BN + exact GELU + Wa2 dot + 3-way softmax -> scores[b][3] ----------
__global__ __launch_bounds__(256) void k_att_score(const float* __restrict__ Y,
                                                   const float* __restrict__ meanA,
                                                   const float* __restrict__ rstdA,
                                                   const float* __restrict__ ga,
                                                   const float* __restrict__ bba,
                                                   const float* __restrict__ Wa2,
                                                   const float* __restrict__ ba2,
                                                   float* __restrict__ scores) {
  const int t = threadIdx.x, lane = t & 31, wid = t >> 5;
  const int b = blockIdx.x * 8 + wid;            // one wave32 per batch row
  float s[3];
#pragma unroll
  for (int f = 0; f < 3; ++f) {
    float acc = 0.f;
#pragma unroll
    for (int i = 0; i < 4; ++i) {
      const int ch = lane + 32 * i;
      float v = Y[(size_t)f * B_ROWS * 128 + (size_t)b * 128 + ch];
      v = (v - meanA[f * 128 + ch]) * rstdA[f * 128 + ch] * ga[ch] + bba[ch];
      v = 0.5f * v * (1.f + erff(v * 0.70710678118654752f));   // exact GELU
      acc += v * Wa2[ch];
    }
    for (int off = 16; off > 0; off >>= 1) acc += __shfl_xor(acc, off, 32);
    s[f] = acc + ba2[0];
  }
  if (lane == 0) {
    const float mx = fmaxf(s[0], fmaxf(s[1], s[2]));
    const float e0 = __expf(s[0] - mx), e1 = __expf(s[1] - mx), e2 = __expf(s[2] - mx);
    const float inv = 1.f / (e0 + e1 + e2);
    scores[(size_t)b * 3 + 0] = e0 * inv;
    scores[(size_t)b * 3 + 1] = e1 * inv;
    scores[(size_t)b * 3 + 2] = e2 * inv;
  }
}

// --- K5: fuse = [sc*ctx | sb*body | sf*face] @ W1^T + b1  (K=2304) ----------
__global__ __launch_bounds__(256) void k_gemm_fuse(const float* __restrict__ ctx,
                                                   const float* __restrict__ xbody,
                                                   const float* __restrict__ xface,
                                                   const float* __restrict__ scores,
                                                   const float* __restrict__ W1,
                                                   const float* __restrict__ b1,
                                                   float* __restrict__ fuse) {
  __shared__ __align__(16) bf16 Ahi[64 * T_LD];
  __shared__ __align__(16) bf16 Alo[64 * T_LD];
  __shared__ __align__(16) bf16 Bhi[64 * T_LD];
  __shared__ __align__(16) bf16 Blo[64 * T_LD];
  const int t = threadIdx.x, lane = t & 31, wid = t >> 5;
  const int wm = wid & 3, wn = wid >> 2;
  const int tileN = blockIdx.x * 64, tileM = blockIdx.y * 64;
  const int m = t >> 2, kq = (t & 3) << 3;
  const int row = tileM + m;
  v8f acc0 = {}, acc1 = {};
  for (int kb0 = 0; kb0 < 3 * NF; kb0 += 32) {
    const int seg = kb0 / NF;                 // K-chunks never straddle segments
    const int kkb = kb0 - seg * NF;
    const float sc = scores[(size_t)row * 3 + seg];   // per-row score folded into A
    __syncthreads();
#pragma unroll
    for (int i = 0; i < 8; ++i) {
      const int kk = kkb + kq + i;
      float v;
      if (seg == 0)      v = ctx[(size_t)row * NF + kk];
      else if (seg == 1) v = xbody[(size_t)row * NF + kk];
      else               v = xface[(size_t)row * DF + kk / 3];
      split_store(&Ahi[m * T_LD + kq + i], &Alo[m * T_LD + kq + i], v * sc);
    }
    const float* bp = W1 + (size_t)(tileN + m) * (3 * NF) + kb0 + kq;
#pragma unroll
    for (int i = 0; i < 8; ++i)
      split_store(&Bhi[m * T_LD + kq + i], &Blo[m * T_LD + kq + i], bp[i]);
    __syncthreads();
    wmma_tile(Ahi, Alo, Bhi, Blo, wm, wn, lane, acc0, acc1);
  }
  store_tile(fuse, 256, tileM, tileN, wm, wn, lane, acc0, acc1, b1);
}

// --- K7: out = relu(BN(fuse)) @ Wcat^T + bcat   [256 -> 26] -----------------
__global__ __launch_bounds__(256) void k_final(const float* __restrict__ fuse,
                                               const float* __restrict__ meanF,
                                               const float* __restrict__ rstdF,
                                               const float* __restrict__ g1,
                                               const float* __restrict__ be1,
                                               const float* __restrict__ Wcat,
                                               const float* __restrict__ bcat,
                                               float* __restrict__ out) {
  __shared__ float WcatT[256 * 28];   // transposed, padded stride 28
  __shared__ float hbuf[8 * 256];     // 8 normalized rows
  const int t = threadIdx.x;
  for (int idx = t; idx < 26 * 256; idx += 256) {
    const int j = idx >> 8, n = idx & 255;
    WcatT[n * 28 + j] = Wcat[idx];
  }
  const int bl = t >> 5;                   // local row 0..7
  const int row = blockIdx.x * 8 + bl;
  const int nb = (t & 31) * 8;
#pragma unroll
  for (int i = 0; i < 8; ++i) {
    const int n = nb + i;
    float v = fuse[(size_t)row * 256 + n];
    v = (v - meanF[n]) * rstdF[n] * g1[n] + be1[n];
    hbuf[bl * 256 + n] = fmaxf(v, 0.f);
  }
  __syncthreads();
  const int j = t & 31;
  if (j < 26) {
    float acc = bcat[j];
    for (int n = 0; n < 256; ++n)
      acc += hbuf[bl * 256 + n] * WcatT[n * 28 + j];  // hbuf read is a broadcast
    out[(size_t)row * 26 + j] = acc;
  }
}

// ---------------------------------------------------------------------------
extern "C" void kernel_launch(void* const* d_in, const int* in_sizes, int n_in,
                              void* d_out, int out_size, void* d_ws, size_t ws_size,
                              hipStream_t stream) {
  const float* x_context = (const float*)d_in[0];
  const float* x_body    = (const float*)d_in[1];
  const float* x_face    = (const float*)d_in[2];
  const float* Wc        = (const float*)d_in[3];
  const float* bc        = (const float*)d_in[4];
  const float* Wa1       = (const float*)d_in[5];
  const float* ba1       = (const float*)d_in[6];
  const float* ga        = (const float*)d_in[7];
  const float* bba       = (const float*)d_in[8];
  const float* Wa2       = (const float*)d_in[9];
  const float* ba2       = (const float*)d_in[10];
  const float* W1        = (const float*)d_in[11];
  const float* b1        = (const float*)d_in[12];
  const float* g1        = (const float*)d_in[13];
  const float* be1       = (const float*)d_in[14];
  const float* Wcat      = (const float*)d_in[15];
  const float* bcat      = (const float*)d_in[16];

  float* ws     = (float*)d_ws;
  float* ctx    = ws;                                   // [B, 768]
  float* Y      = ctx + (size_t)B_ROWS * NF;            // [3, B, 128]
  float* scores = Y + (size_t)3 * B_ROWS * 128;         // [B, 3]
  float* meanA  = scores + (size_t)B_ROWS * 3;          // [384]
  float* rstdA  = meanA + 384;                          // [384]
  float* fusep  = rstdA + 384;                          // [B, 256]
  float* meanF  = fusep + (size_t)B_ROWS * 256;         // [256]
  float* rstdF  = meanF + 256;                          // [256]

  k_gemm_ctx<<<dim3(NF / 64, B_ROWS / 64), 256, 0, stream>>>(x_context, Wc, bc, ctx);
  k_gemm_att<<<dim3(128 / 64, B_ROWS / 64, 3), 256, 0, stream>>>(ctx, x_body, x_face,
                                                                 Wa1, ba1, Y);
  k_stats<<<384, 256, 0, stream>>>(Y, 128, meanA, rstdA);
  k_att_score<<<B_ROWS / 8, 256, 0, stream>>>(Y, meanA, rstdA, ga, bba, Wa2, ba2, scores);
  k_gemm_fuse<<<dim3(256 / 64, B_ROWS / 64), 256, 0, stream>>>(ctx, x_body, x_face,
                                                               scores, W1, b1, fusep);
  k_stats<<<256, 256, 0, stream>>>(fusep, 256, meanF, rstdF);
  k_final<<<B_ROWS / 8, 256, 0, stream>>>(fusep, meanF, rstdF, g1, be1, Wcat, bcat,
                                          (float*)d_out);
}
